// RWKVBlock_45913200394609
// MI455X (gfx1250) — matrix operand order repaired and checked by reading, not compiled, not run on hardware
//
#include <hip/hip_runtime.h>
#include <hip/hip_bf16.h>

typedef _Float16 v8h  __attribute__((ext_vector_type(8)));
typedef _Float16 v16h __attribute__((ext_vector_type(16)));
typedef float    v8f  __attribute__((ext_vector_type(8)));
typedef int      v4i  __attribute__((ext_vector_type(4)));

#define BB 8
#define TT 4096
#define DD 512
#define HH 2048
#define NN (BB * TT)

// GEMM block tile
#define BM 128
#define BN 64
#define BKT 32
#define LDA 40   // padded halfs per A-tile row (80B = 20 banks; breaks conflicts)
#define LDB 40   // padded halfs per B-tile column

#ifndef __has_builtin
#define __has_builtin(x) 0
#endif

#if __has_builtin(__builtin_amdgcn_global_load_async_to_lds_b128) && \
    __has_builtin(__builtin_amdgcn_s_wait_asynccnt)
#define USE_ASYNC_LDS 1
#else
#define USE_ASYNC_LDS 0
#endif

// 16B global -> LDS, async (ASYNCcnt) when available, else via VGPRs (ds_store_b128)
__device__ __forceinline__ void copy16_to_lds(_Float16* ldst, const _Float16* gsrc) {
#if USE_ASYNC_LDS
  __builtin_amdgcn_global_load_async_to_lds_b128(
      (__attribute__((address_space(1))) v4i*)gsrc,
      (__attribute__((address_space(3))) v4i*)ldst, 0, 0);
#else
  *(v8h*)ldst = *(const v8h*)gsrc;
#endif
}

template <short N>
__device__ __forceinline__ void wait_async_le() {
#if USE_ASYNC_LDS
  __builtin_amdgcn_s_wait_asynccnt(N);  // constant operand required
#endif
}

// ---------------------------------------------------------------------------
// fp32 W[K, Nout] (row-major) -> f16 transposed WT[Nout, K]
// ---------------------------------------------------------------------------
__global__ void wconv_kernel(const float* __restrict__ W, _Float16* __restrict__ WT,
                             int K, int Nout) {
  int idx = blockIdx.x * blockDim.x + threadIdx.x;
  if (idx >= K * Nout) return;
  int k = idx / Nout, n = idx % Nout;
  WT[(size_t)n * K + k] = (_Float16)W[idx];
}

// ---------------------------------------------------------------------------
// LayerNorm: one wave per row of D=512, shuffle reduction.
// ---------------------------------------------------------------------------
__global__ __launch_bounds__(256) void ln_kernel(const float* __restrict__ x,
                                                 const float* __restrict__ g,
                                                 const float* __restrict__ b,
                                                 float* __restrict__ out, int rows) {
  int wave = threadIdx.x >> 5;
  int lane = threadIdx.x & 31;
  int row = blockIdx.x * 8 + wave;
  if (row >= rows) return;
  const float* xr = x + (size_t)row * DD;
  float vals[16];
  float s = 0.f, s2 = 0.f;
#pragma unroll
  for (int i = 0; i < 16; i++) {
    float v = xr[lane + i * 32];
    vals[i] = v; s += v; s2 += v * v;
  }
#pragma unroll
  for (int o = 16; o >= 1; o >>= 1) {
    s  += __shfl_xor(s,  o, 32);
    s2 += __shfl_xor(s2, o, 32);
  }
  float mean = s * (1.0f / DD);
  float var  = s2 * (1.0f / DD) - mean * mean;
  float rstd = rsqrtf(var + 1e-5f);
  float* orow = out + (size_t)row * DD;
#pragma unroll
  for (int i = 0; i < 16; i++) {
    int c = lane + i * 32;
    orow[c] = (vals[i] - mean) * rstd * g[c] + b[c];
  }
}

// ---------------------------------------------------------------------------
// time_shift -> f16 WMMA A operand
// ---------------------------------------------------------------------------
__global__ void shift_kernel(const float* __restrict__ h, const float* __restrict__ ratio,
                             _Float16* __restrict__ out) {
  size_t idx = (size_t)blockIdx.x * blockDim.x + threadIdx.x;  // < NN*DD
  int c = (int)(idx % DD);
  int t = (int)((idx / DD) % TT);
  float r = fminf(fmaxf(ratio[c], 0.f), 1.f);
  float cur = h[idx];
  float prev = (t == 0) ? 0.f : h[idx - DD];
  out[idx] = (_Float16)(cur * r + prev * (1.f - r));
}

// ---------------------------------------------------------------------------
// RWKV recurrence: one thread per (batch, channel), sequential in T, fp32.
// ---------------------------------------------------------------------------
__global__ void recur_kernel(float* __restrict__ krec, const float* __restrict__ v,
                             const float* __restrict__ td, const float* __restrict__ tf) {
  int idx = blockIdx.x * blockDim.x + threadIdx.x;  // < BB*DD
  int b = idx / DD, c = idx % DD;
  float decay = 1.f / (1.f + expf(-td[c]));
  float bonus = tanhf(tf[c]);
  float state = 0.f;
  size_t base = (size_t)b * TT * DD + c;
  for (int t = 0; t < TT; t++) {
    size_t p = base + (size_t)t * DD;
    float kt = krec[p];
    float vt = v[p];
    float gate = 1.f / (1.f + expf(-(bonus + kt)));
    float o = gate * vt + (1.f - gate) * state;
    state = decay * state + (1.f - decay) * vt;
    krec[p] = o;
  }
}

// tanh(r) * rec -> f16 A operand
__global__ void amul_kernel(const float* __restrict__ r, const float* __restrict__ rec,
                            _Float16* __restrict__ out) {
  size_t idx = (size_t)blockIdx.x * blockDim.x + threadIdx.x;
  out[idx] = (_Float16)(tanhf(r[idx]) * rec[idx]);
}

// ---------------------------------------------------------------------------
// LDS-tiled WMMA GEMM: C[M,Nout] = A[M,K](f16) x WT[Nout,K](f16, pre-transposed)
// Block: 256 thr / 8 waves; 128x64 output tile; waves 4x2, each 32x32 (2x2 wmma).
// K-slabs of 32 double-buffered in LDS via GLOBAL_LOAD_ASYNC_TO_LDS_B128
// (ASYNCcnt, s_wait_asynccnt) or ds_store fallback.
// Fragment layouts per ISA 7.12.2 (wave32):
//   A: lane<16 -> m=lane, K {0..7,16..23}; lane>=16 -> m=lane-16, K {8..15,24..31}
//   B: lane<16 -> n=lane, K {0..7,16..23}; lane>=16 -> n=lane-16, K {8..15,24..31}
//   C: vgpr j -> row j + 8*(lane>=16), col lane&15
// MODE: 0 = +bias; 1 = res+0.5*(acc+bias); 2 = geluExact(acc+bias)->f16;
//       3 = res+0.5*tanh(gate)*(acc+bias)
// ---------------------------------------------------------------------------
template <int MODE>
__global__ __launch_bounds__(256) void gemm_wmma_kernel(
    const _Float16* __restrict__ A, const _Float16* __restrict__ WT,
    const float* __restrict__ bias, float* __restrict__ Cf,
    _Float16* __restrict__ Ch, const float* __restrict__ res,
    const float* __restrict__ gate, int M, int K, int Nout) {
  __shared__ _Float16 Abuf[2][BM * LDA];  // 2 x 10240B
  __shared__ _Float16 Bbuf[2][BN * LDB];  // 2 x  5120B

  const int tid  = threadIdx.x;
  const int wave = tid >> 5;
  const int lane = tid & 31;
  const int l15  = lane & 15;
  const int lh   = lane >> 4;
  const int wm   = wave >> 1;  // 0..3 -> 32-row stripe
  const int wn   = wave & 1;   // 0..1 -> 32-col stripe
  const int row0 = blockIdx.y * BM;
  const int col0 = blockIdx.x * BN;

  // per-thread staging assignment (3 x 16B chunks per thread per K-slab)
  const int ar0 = tid >> 2, ac0 = (tid & 3) * 8;          // A chunk 0
  const int ar1 = (tid + 256) >> 2, ac1 = (tid & 3) * 8;  // A chunk 1
  const int bc  = tid >> 2, bk = (tid & 3) * 8;           // B chunk

  auto issue = [&](int buf, int k0) {
    copy16_to_lds(&Abuf[buf][ar0 * LDA + ac0],
                  A + (size_t)(row0 + ar0) * K + k0 + ac0);
    copy16_to_lds(&Abuf[buf][ar1 * LDA + ac1],
                  A + (size_t)(row0 + ar1) * K + k0 + ac1);
    copy16_to_lds(&Bbuf[buf][bc * LDB + bk],
                  WT + (size_t)(col0 + bc) * K + k0 + bk);
  };

  v8f acc[2][2];
#pragma unroll
  for (int i = 0; i < 2; i++)
#pragma unroll
    for (int j = 0; j < 2; j++) acc[i][j] = (v8f){0.f, 0.f, 0.f, 0.f, 0.f, 0.f, 0.f, 0.f};

  const int nk = K / BKT;
  issue(0, 0);
  for (int kt = 0; kt < nk; ++kt) {
    const int cur = kt & 1;
    const bool has_next = (kt + 1) < nk;
    if (has_next) {
      issue(cur ^ 1, (kt + 1) * BKT);
      wait_async_le<3>();  // current slab's 3 ops retired; next 3 in flight
    } else {
      wait_async_le<0>();
    }
    __syncthreads();

    const _Float16* As = &Abuf[cur][0];
    const _Float16* Bs = &Bbuf[cur][0];
    v16h af[2], bf[2];
#pragma unroll
    for (int i = 0; i < 2; i++) {
      const _Float16* p = As + (size_t)(wm * 32 + i * 16 + l15) * LDA + lh * 8;
      v8h a0 = *(const v8h*)(p);
      v8h a1 = *(const v8h*)(p + 16);
      af[i] = __builtin_shufflevector(a0, a1, 0, 1, 2, 3, 4, 5, 6, 7,
                                      8, 9, 10, 11, 12, 13, 14, 15);
    }
#pragma unroll
    for (int j = 0; j < 2; j++) {
      const _Float16* p = Bs + (size_t)(wn * 32 + j * 16 + l15) * LDB + lh * 8;
      v8h b0 = *(const v8h*)(p);
      v8h b1 = *(const v8h*)(p + 16);
      bf[j] = __builtin_shufflevector(b0, b1, 0, 1, 2, 3, 4, 5, 6, 7,
                                      8, 9, 10, 11, 12, 13, 14, 15);
    }
    __syncthreads();  // protect this buffer: it is refilled next iteration

#pragma unroll
    for (int i = 0; i < 2; i++)
#pragma unroll
      for (int j = 0; j < 2; j++)
        acc[i][j] = __builtin_amdgcn_wmma_f32_16x16x32_f16(
            false, af[i], false, bf[j], (short)0, acc[i][j], false, false);
  }

#pragma unroll
  for (int i = 0; i < 2; i++) {
#pragma unroll
    for (int j = 0; j < 2; j++) {
      int n = col0 + wn * 32 + j * 16 + l15;
      float bcol = bias[n];
      int mb = row0 + wm * 32 + i * 16 + lh * 8;
#pragma unroll
      for (int jj = 0; jj < 8; jj++) {
        size_t idx = (size_t)(mb + jj) * Nout + n;
        float val = acc[i][j][jj] + bcol;
        if constexpr (MODE == 0) {
          Cf[idx] = val;
        } else if constexpr (MODE == 1) {
          Cf[idx] = res[idx] + 0.5f * val;
        } else if constexpr (MODE == 2) {
          Ch[idx] = (_Float16)(0.5f * val * (1.f + erff(val * 0.70710678118f)));
        } else {
          Cf[idx] = res[idx] + 0.5f * (tanhf(gate[idx]) * val);
        }
      }
    }
  }
}

// ---------------------------------------------------------------------------
extern "C" void kernel_launch(void* const* d_in, const int* in_sizes, int n_in,
                              void* d_out, int out_size, void* d_ws, size_t ws_size,
                              hipStream_t stream) {
  const float* x    = (const float*)d_in[0];
  const float* Wk   = (const float*)d_in[1];  const float* bk  = (const float*)d_in[2];
  const float* Wv   = (const float*)d_in[3];  const float* bv  = (const float*)d_in[4];
  const float* Wr   = (const float*)d_in[5];  const float* br  = (const float*)d_in[6];
  const float* Wo   = (const float*)d_in[7];  const float* bo  = (const float*)d_in[8];
  const float* Wck  = (const float*)d_in[9];  const float* bck = (const float*)d_in[10];
  const float* Wcv  = (const float*)d_in[11]; const float* bcv = (const float*)d_in[12];
  const float* Wcr  = (const float*)d_in[13]; const float* bcr = (const float*)d_in[14];
  const float* ln1g = (const float*)d_in[15]; const float* ln1b= (const float*)d_in[16];
  const float* ln2g = (const float*)d_in[17]; const float* ln2b= (const float*)d_in[18];
  const float* td   = (const float*)d_in[19]; const float* tf  = (const float*)d_in[20];
  const float* tmk  = (const float*)d_in[21]; const float* tmv = (const float*)d_in[22];
  const float* tmr  = (const float*)d_in[23]; const float* cmk = (const float*)d_in[24];
  const float* cmr  = (const float*)d_in[25];
  float* out = (float*)d_out;

  char* ws = (char*)d_ws;
  const size_t KB = 1024, MB = 1024 * 1024;
  _Float16* wk16  = (_Float16*)(ws + 0);
  _Float16* wv16  = (_Float16*)(ws + 512 * KB);
  _Float16* wr16  = (_Float16*)(ws + 1024 * KB);
  _Float16* wo16  = (_Float16*)(ws + 1536 * KB);
  _Float16* wcr16 = (_Float16*)(ws + 2048 * KB);
  _Float16* wck16 = (_Float16*)(ws + 2560 * KB);  // [2048,512]  2 MB
  _Float16* wcv16 = (_Float16*)(ws + 4608 * KB);  // [512,2048]  2 MB
  float*    hbuf = (float*)   (ws + 8   * MB);  // h (LN1) -> h2 (LN2)      64 MB
  _Float16* a16  = (_Float16*)(ws + 72  * MB);  // shared f16 A operand     32 MB
  float*    kbuf = (float*)   (ws + 104 * MB);  // k -> rec (in place)      64 MB
  float*    vbuf = (float*)   (ws + 168 * MB);  // v -> x1                  64 MB
  float*    rbuf = (float*)   (ws + 232 * MB);  // r -> cr                  64 MB
  _Float16* g16  = (_Float16*)(ws + 296 * MB);  // gelu(ck) f16 [N,2048]   128 MB

  const int TPB = 256;
  const int nElemBlocks = (NN * DD) / TPB;  // 65536
  dim3 gD(DD / BN, NN / BM);                // (8, 256)
  dim3 gH(HH / BN, NN / BM);                // (32, 256)

  // ---- weight conversion (f16, transposed) ----
  wconv_kernel<<<(DD * DD) / TPB, TPB, 0, stream>>>(Wk,  wk16,  DD, DD);
  wconv_kernel<<<(DD * DD) / TPB, TPB, 0, stream>>>(Wv,  wv16,  DD, DD);
  wconv_kernel<<<(DD * DD) / TPB, TPB, 0, stream>>>(Wr,  wr16,  DD, DD);
  wconv_kernel<<<(DD * DD) / TPB, TPB, 0, stream>>>(Wo,  wo16,  DD, DD);
  wconv_kernel<<<(DD * DD) / TPB, TPB, 0, stream>>>(Wcr, wcr16, DD, DD);
  wconv_kernel<<<(DD * HH) / TPB, TPB, 0, stream>>>(Wck, wck16, DD, HH);
  wconv_kernel<<<(HH * DD) / TPB, TPB, 0, stream>>>(Wcv, wcv16, HH, DD);

  // ---- time mixing ----
  ln_kernel<<<NN / 8, TPB, 0, stream>>>(x, ln1g, ln1b, hbuf, NN);

  shift_kernel<<<nElemBlocks, TPB, 0, stream>>>(hbuf, tmk, a16);
  gemm_wmma_kernel<0><<<gD, TPB, 0, stream>>>(a16, wk16, bk, kbuf, nullptr,
                                              nullptr, nullptr, NN, DD, DD);
  shift_kernel<<<nElemBlocks, TPB, 0, stream>>>(hbuf, tmv, a16);
  gemm_wmma_kernel<0><<<gD, TPB, 0, stream>>>(a16, wv16, bv, vbuf, nullptr,
                                              nullptr, nullptr, NN, DD, DD);
  shift_kernel<<<nElemBlocks, TPB, 0, stream>>>(hbuf, tmr, a16);
  gemm_wmma_kernel<0><<<gD, TPB, 0, stream>>>(a16, wr16, br, rbuf, nullptr,
                                              nullptr, nullptr, NN, DD, DD);

  recur_kernel<<<(BB * DD) / TPB, TPB, 0, stream>>>(kbuf, vbuf, td, tf);
  amul_kernel<<<nElemBlocks, TPB, 0, stream>>>(rbuf, kbuf, a16);
  // x1 = x + 0.5*(A @ Wo + bo) -> vbuf
  gemm_wmma_kernel<1><<<gD, TPB, 0, stream>>>(a16, wo16, bo, vbuf, nullptr,
                                              x, nullptr, NN, DD, DD);

  // ---- channel mixing ----
  ln_kernel<<<NN / 8, TPB, 0, stream>>>(vbuf, ln2g, ln2b, hbuf, NN);

  shift_kernel<<<nElemBlocks, TPB, 0, stream>>>(hbuf, cmr, a16);
  gemm_wmma_kernel<0><<<gD, TPB, 0, stream>>>(a16, wcr16, bcr, rbuf, nullptr,
                                              nullptr, nullptr, NN, DD, DD);

  shift_kernel<<<nElemBlocks, TPB, 0, stream>>>(hbuf, cmk, a16);
  gemm_wmma_kernel<2><<<gH, TPB, 0, stream>>>(a16, wck16, bck, nullptr, g16,
                                              nullptr, nullptr, NN, DD, HH);
  // out = x1 + 0.5 * tanh(cr) * (gelu(ck) @ Wcv + bcv)
  gemm_wmma_kernel<3><<<gD, TPB, 0, stream>>>(g16, wcv16, bcv, out, nullptr,
                                              vbuf, rbuf, NN, HH, DD);
}
